// NemotronFlashMoE_89850715833066
// MI455X (gfx1250) — matrix operand order repaired and checked by reading, not compiled, not run on hardware
//
#include <hip/hip_runtime.h>
#include <hip/hip_bf16.h>

#define T_TOK 8192
#define HDIM  2048
#define NEXP  8
#define IDIM  1024

typedef __attribute__((ext_vector_type(16))) __bf16 v16bf;
typedef __attribute__((ext_vector_type(8)))  __bf16 v8bf;
typedef __attribute__((ext_vector_type(8)))  float  v8f;

// ---------------------------------------------------------------- conversion
__global__ void cvt_f32_bf16_kernel(const float* __restrict__ in,
                                    __bf16* __restrict__ out, int n8) {
  int i = blockIdx.x * blockDim.x + threadIdx.x;
  if (i >= n8) return;
  const float4* p = (const float4*)in + (size_t)i * 2;
  float4 a = p[0], b = p[1];
  v8bf o;
  o[0] = (__bf16)a.x; o[1] = (__bf16)a.y; o[2] = (__bf16)a.z; o[3] = (__bf16)a.w;
  o[4] = (__bf16)b.x; o[5] = (__bf16)b.y; o[6] = (__bf16)b.z; o[7] = (__bf16)b.w;
  ((v8bf*)out)[i] = o;
}

// ---------------------------------------------------------------- router
__global__ void router_kernel(const float* __restrict__ x,
                              const float* __restrict__ wg,
                              float* __restrict__ topw, int* __restrict__ topi,
                              int* __restrict__ counts) {
  int wave = threadIdx.x >> 5, lane = threadIdx.x & 31;
  int t = blockIdx.x * 8 + wave;
  if (t >= T_TOK) return;
  const float* xr = x + (size_t)t * HDIM;
  float acc[NEXP];
#pragma unroll
  for (int e = 0; e < NEXP; e++) acc[e] = 0.f;
  for (int h = lane; h < HDIM; h += 32) {
    float xv = xr[h];
#pragma unroll
    for (int e = 0; e < NEXP; e++) acc[e] += xv * wg[e * HDIM + h];
  }
#pragma unroll
  for (int e = 0; e < NEXP; e++) {
#pragma unroll
    for (int off = 16; off > 0; off >>= 1) acc[e] += __shfl_xor(acc[e], off, 32);
  }
  if (lane == 0) {
    int a = 0;
#pragma unroll
    for (int e = 1; e < NEXP; e++) if (acc[e] > acc[a]) a = e;
    int b = (a == 0) ? 1 : 0;
#pragma unroll
    for (int e = 0; e < NEXP; e++) if (e != a && acc[e] > acc[b]) b = e;
    float wa = 1.f / (1.f + __expf(acc[b] - acc[a])); // softmax denom cancels
    topw[t * 2 + 0] = wa;       topw[t * 2 + 1] = 1.f - wa;
    topi[t * 2 + 0] = a;        topi[t * 2 + 1] = b;
    atomicAdd(&counts[a], 1);   atomicAdd(&counts[b], 1);
  }
}

__global__ void scan_kernel(const int* __restrict__ counts, int* __restrict__ offsets) {
  if (threadIdx.x == 0 && blockIdx.x == 0) {
    int s = 0;
#pragma unroll
    for (int e = 0; e < NEXP; e++) { offsets[e] = s; s += counts[e]; }
  }
}

__global__ void build_kernel(const int* __restrict__ topi,
                             const int* __restrict__ offsets,
                             int* __restrict__ cursor, int* __restrict__ rows) {
  int t = blockIdx.x * blockDim.x + threadIdx.x;
  if (t >= T_TOK) return;
#pragma unroll
  for (int k = 0; k < 2; k++) {
    int e = topi[t * 2 + k];
    int p = atomicAdd(&cursor[e], 1);
    rows[offsets[e] + p] = t * 2 + k;   // encodes (token, slot)
  }
}

// ---------------------------------------------------------------- WMMA helpers
template <bool RA>
static __device__ __forceinline__ v8f wmma_bf16(v16bf a, v16bf b, v8f c) {
  // reuse_a=RA: A operand identical to the previous (identical) WMMA -> cache it
  return __builtin_amdgcn_wmma_f32_16x16x32_bf16(false, a, false, b,
                                                 (short)0, c, RA, false);
}

// A fragment (16x32 bf16): lane L<16 holds M=L, K in {0..15};
// lane L>=16 holds M=L-16, K in {16..31}  -> two 16B LDS loads.
static __device__ __forceinline__ v16bf frag_a(const __bf16* row, int lhalf) {
  v8bf lo = *(const v8bf*)(row + lhalf * 16);
  v8bf hi = *(const v8bf*)(row + lhalf * 16 + 8);
  v16bf r;
#pragma unroll
  for (int i = 0; i < 8; i++) { r[i] = lo[i]; r[i + 8] = hi[i]; }
  return r;
}

static __device__ __forceinline__ float fast_silu_mul(float g, float u) {
  // g * sigmoid(g) * u with v_rcp_f32 instead of the IEEE division expansion
  float s = __builtin_amdgcn_rcpf(1.f + __expf(-g));
  return g * s * u;
}

#define LDW 48   // LDS row stride in bf16 (96B: 32B-aligned, conflict-skewed)

// ---------------------------------------------------------------- up (gate+up) GEMM
__launch_bounds__(256, 2)
__global__ void moe_up_kernel(const __bf16* __restrict__ xb,
                              const __bf16* __restrict__ w1b,
                              const __bf16* __restrict__ w3b,
                              const int* __restrict__ counts,
                              const int* __restrict__ offsets,
                              const int* __restrict__ rows,
                              __bf16* __restrict__ hb) {
  const int e   = blockIdx.z;
  const int cnt = counts[e];
  const int m0  = blockIdx.y * 64;
  if (m0 >= cnt) return;
  const int base = offsets[e];
  const int n0   = blockIdx.x * 128;

  __shared__ __align__(32) __bf16 lA [2][64 * LDW];
  __shared__ __align__(32) __bf16 lB1[2][128 * LDW];
  __shared__ __align__(32) __bf16 lB3[2][128 * LDW];
  __shared__ int toks[64];

  const int tid = threadIdx.x;
  if (tid < 64) {
    int mr = m0 + tid; if (mr >= cnt) mr = cnt - 1;
    toks[tid] = rows[base + mr] >> 1;
  }
  __syncthreads();

  const int wave = tid >> 5, lane = tid & 31;
  const int wm = wave >> 1, wn = wave & 1;          // 4 (M) x 2 (N) wave grid
  const int lrow = lane & 15, lhalf = lane >> 4;

  v8f acc1[4] = {}; v8f acc3[4] = {};

  const int arow = tid >> 2, aq = tid & 3;          // A: 64 rows x 4x16B
  const int brow = tid >> 1, bh = tid & 1;          // B: 128 rows x 2x32B
  const __bf16* xrow  = xb  + (size_t)toks[arow] * HDIM + aq * 8;
  const __bf16* w1row = w1b + ((size_t)e * IDIM + n0 + brow) * HDIM + bh * 16;
  const __bf16* w3row = w3b + ((size_t)e * IDIM + n0 + brow) * HDIM + bh * 16;

  const int aoff = arow * LDW + aq * 8;
  const int boff = brow * LDW + bh * 16;

  v8bf sA, sB1a, sB1b, sB3a, sB3b;                  // register staging
  auto load_stage = [&](int k0) {
    sA   = *(const v8bf*)(xrow  + k0);
    sB1a = *(const v8bf*)(w1row + k0);
    sB1b = *(const v8bf*)(w1row + k0 + 8);
    sB3a = *(const v8bf*)(w3row + k0);
    sB3b = *(const v8bf*)(w3row + k0 + 8);
  };
  auto store_stage = [&](int b) {
    *(v8bf*)&lA [b][aoff]     = sA;
    *(v8bf*)&lB1[b][boff]     = sB1a;
    *(v8bf*)&lB1[b][boff + 8] = sB1b;
    *(v8bf*)&lB3[b][boff]     = sB3a;
    *(v8bf*)&lB3[b][boff + 8] = sB3b;
  };

  load_stage(0);
  store_stage(0);
  __syncthreads();

  int buf = 0;
  for (int k0 = 32; k0 <= HDIM; k0 += 32) {
    const bool more = (k0 < HDIM);
    if (more) load_stage(k0);                        // overlap with compute below

    v16bf af = frag_a(&lA[buf][(wm * 16 + lrow) * LDW], lhalf);
    const __bf16* pb1 = &lB1[buf][(wn * 64 + lrow) * LDW + lhalf * 16];
    const __bf16* pb3 = &lB3[buf][(wn * 64 + lrow) * LDW + lhalf * 16];

#define UP_PAIR(J, RA0)                                                    \
    { v16bf b1 = *(const v16bf*)(pb1 + (J) * 16 * LDW);                    \
      acc1[J] = wmma_bf16<RA0>(af, b1, acc1[J]);                           \
      v16bf b3 = *(const v16bf*)(pb3 + (J) * 16 * LDW);                    \
      acc3[J] = wmma_bf16<true>(af, b3, acc3[J]); }
    UP_PAIR(0, false)
    UP_PAIR(1, true)
    UP_PAIR(2, true)
    UP_PAIR(3, true)
#undef UP_PAIR

    if (more) store_stage(buf ^ 1);                  // write other buffer
    __syncthreads();
    buf ^= 1;
  }

  // SwiGLU epilogue -> gathered h rows (bf16)
#pragma unroll
  for (int j = 0; j < 4; j++) {
    int n = n0 + wn * 64 + j * 16 + lrow;
#pragma unroll
    for (int v = 0; v < 8; v++) {
      int m = m0 + wm * 16 + v + 8 * lhalf;
      if (m < cnt) {
        float hv = fast_silu_mul(acc1[j][v], acc3[j][v]);
        hb[(size_t)(base + m) * IDIM + n] = (__bf16)hv;
      }
    }
  }
}

// ---------------------------------------------------------------- down GEMM + scatter
__launch_bounds__(256, 2)
__global__ void moe_down_kernel(const __bf16* __restrict__ hb,
                                const __bf16* __restrict__ w2b,
                                const int* __restrict__ counts,
                                const int* __restrict__ offsets,
                                const int* __restrict__ rows,
                                const float* __restrict__ topw,
                                float* __restrict__ out) {
  const int e   = blockIdx.z;
  const int cnt = counts[e];
  const int m0  = blockIdx.y * 64;
  if (m0 >= cnt) return;
  const int base = offsets[e];
  const int n0   = blockIdx.x * 128;

  __shared__ __align__(32) __bf16 lA[2][64 * LDW];
  __shared__ __align__(32) __bf16 lB[2][128 * LDW];
  __shared__ int   toks[64];
  __shared__ float wts[64];

  const int tid = threadIdx.x;
  if (tid < 64) {
    int mr = m0 + tid; if (mr >= cnt) mr = cnt - 1;
    int rc = rows[base + mr];
    toks[tid] = rc >> 1;
    wts[tid]  = topw[rc];
  }
  __syncthreads();

  const int wave = tid >> 5, lane = tid & 31;
  const int wm = wave >> 1, wn = wave & 1;
  const int lrow = lane & 15, lhalf = lane >> 4;

  v8f acc[4] = {};

  const int arow = tid >> 2, aq = tid & 3;
  const int brow = tid >> 1, bh = tid & 1;
  int ar = m0 + arow; if (ar >= cnt) ar = cnt - 1;
  const __bf16* hrow  = hb  + (size_t)(base + ar) * IDIM + aq * 8;
  const __bf16* w2row = w2b + ((size_t)e * HDIM + n0 + brow) * IDIM + bh * 16;

  const int aoff = arow * LDW + aq * 8;
  const int boff = brow * LDW + bh * 16;

  v8bf sA, sBa, sBb;
  auto load_stage = [&](int k0) {
    sA  = *(const v8bf*)(hrow  + k0);
    sBa = *(const v8bf*)(w2row + k0);
    sBb = *(const v8bf*)(w2row + k0 + 8);
  };
  auto store_stage = [&](int b) {
    *(v8bf*)&lA[b][aoff]     = sA;
    *(v8bf*)&lB[b][boff]     = sBa;
    *(v8bf*)&lB[b][boff + 8] = sBb;
  };

  load_stage(0);
  store_stage(0);
  __syncthreads();

  int buf = 0;
  for (int k0 = 32; k0 <= IDIM; k0 += 32) {
    const bool more = (k0 < IDIM);
    if (more) load_stage(k0);

    v16bf af = frag_a(&lA[buf][(wm * 16 + lrow) * LDW], lhalf);
    const __bf16* pb = &lB[buf][(wn * 64 + lrow) * LDW + lhalf * 16];

#define DN_MMA(J, RA0)                                                     \
    { v16bf bfv = *(const v16bf*)(pb + (J) * 16 * LDW);                    \
      acc[J] = wmma_bf16<RA0>(af, bfv, acc[J]); }
    DN_MMA(0, false)
    DN_MMA(1, true)
    DN_MMA(2, true)
    DN_MMA(3, true)
#undef DN_MMA

    if (more) store_stage(buf ^ 1);
    __syncthreads();
    buf ^= 1;
  }

  // weighted scatter-add into fp32 output (each token hit by exactly 2 experts)
#pragma unroll
  for (int j = 0; j < 4; j++) {
    int n = n0 + wn * 64 + j * 16 + lrow;
#pragma unroll
    for (int v = 0; v < 8; v++) {
      int ml = wm * 16 + v + 8 * lhalf;
      int m  = m0 + ml;
      if (m < cnt) {
        float val = acc[j][v] * wts[ml];
        atomicAdd(&out[(size_t)toks[ml] * HDIM + n], val);
      }
    }
  }
}

// ---------------------------------------------------------------- launcher
extern "C" void kernel_launch(void* const* d_in, const int* in_sizes, int n_in,
                              void* d_out, int out_size, void* d_ws, size_t ws_size,
                              hipStream_t stream) {
  const float* x  = (const float*)d_in[0];
  const float* wg = (const float*)d_in[1];
  const float* w1 = (const float*)d_in[2];
  const float* w3 = (const float*)d_in[3];
  const float* w2 = (const float*)d_in[4];
  float* out = (float*)d_out;

  char* ws = (char*)d_ws;
  size_t off = 0;
  auto alloc = [&](size_t bytes) -> char* {
    char* p = ws + off;
    off = (off + bytes + 255) & ~(size_t)255;
    return p;
  };

  __bf16* xb   = (__bf16*)alloc((size_t)T_TOK * HDIM * 2);
  __bf16* w1b  = (__bf16*)alloc((size_t)NEXP * IDIM * HDIM * 2);
  __bf16* w3b  = (__bf16*)alloc((size_t)NEXP * IDIM * HDIM * 2);
  __bf16* w2b  = (__bf16*)alloc((size_t)NEXP * HDIM * IDIM * 2);
  __bf16* hb   = (__bf16*)alloc((size_t)2 * T_TOK * IDIM * 2);
  float*  topw = (float*)alloc((size_t)T_TOK * 2 * 4);
  int*    topi = (int*)  alloc((size_t)T_TOK * 2 * 4);
  int*    rows = (int*)  alloc((size_t)2 * T_TOK * 4);
  int*    ctrl = (int*)  alloc(3 * NEXP * 4);
  int* counts  = ctrl;
  int* cursor  = ctrl + NEXP;
  int* offsets = ctrl + 2 * NEXP;

  hipMemsetAsync(ctrl, 0, 2 * NEXP * sizeof(int), stream);          // counts+cursor
  hipMemsetAsync(out, 0, (size_t)out_size * sizeof(float), stream); // fp32 accum

  int n8;
  n8 = (T_TOK * HDIM) / 8;
  cvt_f32_bf16_kernel<<<(n8 + 255) / 256, 256, 0, stream>>>(x, xb, n8);
  n8 = (NEXP * IDIM * HDIM) / 8;
  cvt_f32_bf16_kernel<<<(n8 + 255) / 256, 256, 0, stream>>>(w1, w1b, n8);
  cvt_f32_bf16_kernel<<<(n8 + 255) / 256, 256, 0, stream>>>(w3, w3b, n8);
  cvt_f32_bf16_kernel<<<(n8 + 255) / 256, 256, 0, stream>>>(w2, w2b, n8);

  router_kernel<<<T_TOK / 8, 256, 0, stream>>>(x, wg, topw, topi, counts);
  scan_kernel<<<1, 32, 0, stream>>>(counts, offsets);
  build_kernel<<<T_TOK / 256, 256, 0, stream>>>(topi, offsets, cursor, rows);

  moe_up_kernel<<<dim3(IDIM / 128, T_TOK / 64, NEXP), 256, 0, stream>>>(
      xb, w1b, w3b, counts, offsets, rows, hb);
  moe_down_kernel<<<dim3(HDIM / 128, T_TOK / 64, NEXP), 256, 0, stream>>>(
      hb, w2b, counts, offsets, rows, topw, out);
}
